// FGCNet_36240934043737
// MI455X (gfx1250) — matrix operand order repaired and checked by reading, not compiled, not run on hardware
//
#include <hip/hip_runtime.h>
#include <hip/hip_bf16.h>
#include <math.h>

// ---------------------------------------------------------------------------
// FeaStConv U-net for MI455X (gfx1250, wave32, WMMA).
// Dense projections + fc head use v_wmma_f32_16x16x32_bf16 with branchless
// b128-vectorized fragment loads. Edge attention / scatter is fp32 VALU +
// global_atomic_add_f32 (HBM bound).
// ---------------------------------------------------------------------------

#define NH 9  // attention heads

typedef __attribute__((ext_vector_type(16))) __bf16       v16bf;
typedef __attribute__((ext_vector_type(8)))  float        v8f;
typedef __attribute__((ext_vector_type(8)))  unsigned int v8u;

union Frag {
  v8u   u;
  v16bf b;
  uint4 q[2];
};

__device__ __forceinline__ unsigned short f32_to_bf16(float f) {
  unsigned int u = __float_as_uint(f);
  u += 0x7FFFu + ((u >> 16) & 1u);  // round-to-nearest-even
  return (unsigned short)(u >> 16);
}

// ---------------- conversion kernels ----------------

// x [n,K] f32 row-major -> dst [n,Kpad] bf16 row-major, zero padded
__global__ void fgc_x_to_bf16(const float* __restrict__ x,
                              unsigned short* __restrict__ dst,
                              int n, int K, int Kpad) {
  long idx = (long)blockIdx.x * blockDim.x + threadIdx.x;
  long tot = (long)n * Kpad;
  if (idx >= tot) return;
  int i = (int)(idx / Kpad);
  int kp = (int)(idx % Kpad);
  dst[idx] = (kp < K) ? f32_to_bf16(x[(long)i * K + kp]) : (unsigned short)0;
}

// W [K,N] f32 row-major -> dst [N,Kpad] bf16 (column-major of W), zero padded
__global__ void fgc_w_to_bf16_cm(const float* __restrict__ W,
                                 unsigned short* __restrict__ dst,
                                 int K, int N, int Kpad) {
  long idx = (long)blockIdx.x * blockDim.x + threadIdx.x;
  long tot = (long)N * Kpad;
  if (idx >= tot) return;
  int ncol = (int)(idx / Kpad);
  int kp = (int)(idx % Kpad);
  dst[idx] = (kp < K) ? f32_to_bf16(W[(long)kp * N + ncol]) : (unsigned short)0;
}

// ---------------- WMMA GEMM: C[M,N] = A[M,Kpad] * B(cm)[N,Kpad] ----------------
// One wave per 16x16 output tile. Fragment layouts per CDNA5 ISA 7.12.2:
//   A 16x32 bf16 : lane m -> row m%16; vgpr i pair at (i<4 ? 2i : 16+2(i-4)) + 8*hi
//                  => two contiguous 16B runs: [8hi..8hi+7], [16+8hi..16+8hi+7]
//   B 32x16 bf16 : lane n -> col n%16; same k layout as A
//                  => two contiguous 16B runs: [8hi..8hi+7], [16+8hi..16+8hi+7]
//   C/D f32      : vgpr i -> row i + 8*hi, col lane%16
// Out-of-range rows are CLAMPED (not predicated) so no EXEC toggling in the
// hot loop; the store guard discards junk rows.
__global__ void fgc_gemm_wmma(const unsigned short* __restrict__ A,
                              const unsigned short* __restrict__ B,
                              float* __restrict__ C,
                              int M, int N, int Kpad) {
  int lane = threadIdx.x & 31;
  int wave = blockIdx.x * (blockDim.x >> 5) + (threadIdx.x >> 5);
  int tilesN = N >> 4;
  int tilesM = (M + 15) >> 4;
  if (wave >= tilesM * tilesN) return;   // wave-uniform: EXEC stays all-ones
  int tM = wave / tilesN;
  int tN = wave % tilesN;
  int lr = lane & 15;
  int hi = lane >> 4;

  int r = tM * 16 + lr;
  if (r >= M) r = M - 1;                 // clamp: branchless, discarded at store
  const unsigned short* arow = A + (size_t)r * Kpad + 8 * hi;
  const unsigned short* bcol = B + (size_t)(tN * 16 + lr) * Kpad + 8 * hi;

  v8f acc = {0.f, 0.f, 0.f, 0.f, 0.f, 0.f, 0.f, 0.f};
  for (int k = 0; k < Kpad; k += 32) {
    Frag a, b;
    a.q[0] = *(const uint4*)(arow + k);
    a.q[1] = *(const uint4*)(arow + k + 16);
    b.q[0] = *(const uint4*)(bcol + k);
    b.q[1] = *(const uint4*)(bcol + k + 16);
    acc = __builtin_amdgcn_wmma_f32_16x16x32_bf16(
        false, a.b, false, b.b, (short)0, acc, false, false);
  }
#pragma unroll
  for (int i = 0; i < 8; ++i) {
    int row = tM * 16 + i + 8 * hi;
    if (row < M) C[(size_t)row * N + tN * 16 + lr] = acc[i];
  }
}

// ---------------- FeaStConv edge kernel (one wave32 per edge) ----------------
__global__ void fgc_feast_edges(const float* __restrict__ x, int cin,
                                const float* __restrict__ xW, int out,
                                const int* __restrict__ src,
                                const int* __restrict__ dst, int E,
                                const float* __restrict__ u,
                                const float* __restrict__ c,
                                float* __restrict__ agg,
                                float* __restrict__ cnt) {
  int lane = threadIdx.x & 31;
  int e = blockIdx.x * (blockDim.x >> 5) + (threadIdx.x >> 5);
  if (e >= E) return;
  int s = src[e];
  int d0 = dst[e];

  float dots[NH];
#pragma unroll
  for (int h = 0; h < NH; ++h) dots[h] = 0.f;
  for (int k = lane; k < cin; k += 32) {
    float diff = x[(long)s * cin + k] - x[(long)d0 * cin + k];
#pragma unroll
    for (int h = 0; h < NH; ++h) dots[h] += diff * u[k * NH + h];
  }
#pragma unroll
  for (int off = 16; off >= 1; off >>= 1) {
#pragma unroll
    for (int h = 0; h < NH; ++h) dots[h] += __shfl_xor(dots[h], off, 32);
  }
  // softmax over heads (replicated per lane; dots already wave-broadcast)
  float q[NH];
  float mx = -3.0e38f;
#pragma unroll
  for (int h = 0; h < NH; ++h) { q[h] = dots[h] + c[h]; mx = fmaxf(mx, q[h]); }
  float ssum = 0.f;
#pragma unroll
  for (int h = 0; h < NH; ++h) { q[h] = __expf(q[h] - mx); ssum += q[h]; }
  float inv = 1.f / ssum;
#pragma unroll
  for (int h = 0; h < NH; ++h) q[h] *= inv;

  const float* xws = xW + (size_t)s * (NH * out);
  for (int o = lane; o < out; o += 32) {
    float m = 0.f;
#pragma unroll
    for (int h = 0; h < NH; ++h) m += q[h] * xws[h * out + o];
    atomicAdd(&agg[(size_t)d0 * out + o], m);
  }
  if (lane == 0) atomicAdd(&cnt[d0], 1.0f);
}

// ---------------- self-loop head weights: qc = softmax(c) (once / layer) -----
__global__ void fgc_softmax_c(const float* __restrict__ c,
                              float* __restrict__ qc) {
  if (threadIdx.x == 0 && blockIdx.x == 0) {
    float e[NH];
    float mx = -3.0e38f;
    for (int h = 0; h < NH; ++h) mx = fmaxf(mx, c[h]);
    float s = 0.f;
    for (int h = 0; h < NH; ++h) { e[h] = __expf(c[h] - mx); s += e[h]; }
    float inv = 1.f / s;
    for (int h = 0; h < NH; ++h) qc[h] = e[h] * inv;
  }
}

// ---------------- FeaStConv finalize: self-loop + mean + bias (+lrelu) -------
__global__ void fgc_feast_finalize(const float* __restrict__ xW,
                                   const float* __restrict__ agg,
                                   const float* __restrict__ cnt,
                                   const float* __restrict__ qc,
                                   const float* __restrict__ b,
                                   float* __restrict__ outbuf,
                                   int n, int out, int ldo, int col0,
                                   int do_lrelu) {
  long idx = (long)blockIdx.x * blockDim.x + threadIdx.x;
  if (idx >= (long)n * out) return;
  int i = (int)(idx / out);
  int o = (int)(idx % out);
  // self-loop attention: x_j - x_i == 0 -> q = softmax(c) (precomputed in qc)
  float selfv = 0.f;
  const float* xwi = xW + (size_t)i * (NH * out);
#pragma unroll
  for (int h = 0; h < NH; ++h) selfv += qc[h] * xwi[h * out + o];
  float v = (agg[(size_t)i * out + o] + selfv) / (cnt[i] + 1.0f) + b[o];
  if (do_lrelu) v = (v > 0.f) ? v : 0.1f * v;
  outbuf[(size_t)i * ldo + col0 + o] = v;
}

// ---------------- pooling / gather / concat helpers ----------------

__global__ void fgc_pool_scatter(const float* __restrict__ x,
                                 const int* __restrict__ clust,
                                 float* __restrict__ psum, int n, int C) {
  long idx = (long)blockIdx.x * blockDim.x + threadIdx.x;
  if (idx >= (long)n * C) return;
  int i = (int)(idx / C);
  int f = (int)(idx % C);
  atomicAdd(&psum[(size_t)clust[i] * C + f], x[idx]);
}

__global__ void fgc_pool_cnt(const int* __restrict__ clust,
                             float* __restrict__ pcnt, int n) {
  int i = blockIdx.x * blockDim.x + threadIdx.x;
  if (i >= n) return;
  atomicAdd(&pcnt[clust[i]], 1.0f);
}

__global__ void fgc_pool_div(float* __restrict__ dst,
                             const float* __restrict__ psum,
                             const float* __restrict__ pcnt, int m, int C) {
  long idx = (long)blockIdx.x * blockDim.x + threadIdx.x;
  if (idx >= (long)m * C) return;
  int j = (int)(idx / C);
  dst[idx] = psum[idx] / fmaxf(pcnt[j], 1.0f);
}

__global__ void fgc_gather_rows(const float* __restrict__ src,
                                const int* __restrict__ idx,
                                float* __restrict__ dst, int n, int C) {
  long t = (long)blockIdx.x * blockDim.x + threadIdx.x;
  if (t >= (long)n * C) return;
  int i = (int)(t / C);
  int f = (int)(t % C);
  dst[t] = src[(size_t)idx[i] * C + f];
}

__global__ void fgc_copy_cols(const float* __restrict__ src,
                              float* __restrict__ dst,
                              int n, int C, int ldo, int col0) {
  long t = (long)blockIdx.x * blockDim.x + threadIdx.x;
  if (t >= (long)n * C) return;
  int i = (int)(t / C);
  int f = (int)(t % C);
  dst[(size_t)i * ldo + col0 + f] = src[t];
}

// ---------------- fused fc head: lrelu(y@fc1+b1)@fc2+b2, row-normalized ------
// One wave per 16-row tile; fc1 via WMMA (K=32, one wmma per 16-col tile),
// fc2 contraction folded into the N-tile loop, single cross-lane reduce.
__global__ void fgc_fc_fused(const unsigned short* __restrict__ yb,   // [n,32] bf16
                             const unsigned short* __restrict__ f1b,  // [1024,32] bf16 cm
                             const float* __restrict__ fb1,           // [1024]
                             const float* __restrict__ w2,            // [1024,3]
                             const float* __restrict__ b2,            // [3]
                             float* __restrict__ out,                 // [n,3]
                             int n) {
  int lane = threadIdx.x & 31;
  int wave = blockIdx.x * (blockDim.x >> 5) + (threadIdx.x >> 5);
  int tilesM = (n + 15) >> 4;
  if (wave >= tilesM) return;   // wave-uniform
  int lr = lane & 15;
  int hi = lane >> 4;

  int r = wave * 16 + lr;
  if (r >= n) r = n - 1;                 // clamp (branchless), guard at store
  const unsigned short* arow = yb + (size_t)r * 32 + 8 * hi;
  Frag a;
  a.q[0] = *(const uint4*)(arow);
  a.q[1] = *(const uint4*)(arow + 16);

  float acc3[8][3];
#pragma unroll
  for (int i = 0; i < 8; ++i) { acc3[i][0] = acc3[i][1] = acc3[i][2] = 0.f; }

  for (int tN = 0; tN < 64; ++tN) {
    int col = tN * 16 + lr;                     // this lane's fc1 output column
    const unsigned short* bcol = f1b + (size_t)col * 32 + 8 * hi;
    Frag b;
    b.q[0] = *(const uint4*)(bcol);
    b.q[1] = *(const uint4*)(bcol + 16);
    v8f z = {0.f, 0.f, 0.f, 0.f, 0.f, 0.f, 0.f, 0.f};
    v8f d = __builtin_amdgcn_wmma_f32_16x16x32_bf16(
        false, a.b, false, b.b, (short)0, z, false, false);
    float bias = fb1[col];
    float w20 = w2[col * 3 + 0], w21 = w2[col * 3 + 1], w22 = w2[col * 3 + 2];
#pragma unroll
    for (int i = 0; i < 8; ++i) {
      float v = d[i] + bias;
      v = (v > 0.f) ? v : 0.1f * v;             // lrelu
      acc3[i][0] += v * w20;
      acc3[i][1] += v * w21;
      acc3[i][2] += v * w22;
    }
  }
  // reduce over the 16 lanes of each half-wave (columns)
#pragma unroll
  for (int m = 1; m < 16; m <<= 1) {
#pragma unroll
    for (int i = 0; i < 8; ++i) {
      acc3[i][0] += __shfl_xor(acc3[i][0], m, 32);
      acc3[i][1] += __shfl_xor(acc3[i][1], m, 32);
      acc3[i][2] += __shfl_xor(acc3[i][2], m, 32);
    }
  }
  if (lr == 0) {
#pragma unroll
    for (int i = 0; i < 8; ++i) {
      int row = wave * 16 + i + 8 * hi;
      if (row < n) {
        float v0 = acc3[i][0] + b2[0];
        float v1 = acc3[i][1] + b2[1];
        float v2 = acc3[i][2] + b2[2];
        float nrm = sqrtf(v0 * v0 + v1 * v1 + v2 * v2);
        float inv = 1.0f / fmaxf(nrm, 1e-12f);
        out[(size_t)row * 3 + 0] = v0 * inv;
        out[(size_t)row * 3 + 1] = v1 * inv;
        out[(size_t)row * 3 + 2] = v2 * inv;
      }
    }
  }
}

// ---------------------------------------------------------------------------
// host orchestration
// ---------------------------------------------------------------------------

static inline int cdiv(long a, long b) { return (int)((a + b - 1) / b); }

static void run_feast(hipStream_t stream,
                      const float* xin, int n, int cin, int out,
                      const int* src, const int* dst, int E,
                      const float* u, const float* c, const float* W,
                      const float* b,
                      unsigned short* xb, unsigned short* Wb, float* xW,
                      float* agg, float* cnt, float* qc,
                      float* outbuf, int ldo, int col0, int lrelu) {
  int Kpad = (cin + 31) & ~31;
  int Ncols = NH * out;
  fgc_x_to_bf16<<<cdiv((long)n * Kpad, 256), 256, 0, stream>>>(xin, xb, n, cin, Kpad);
  fgc_w_to_bf16_cm<<<cdiv((long)Ncols * Kpad, 256), 256, 0, stream>>>(W, Wb, cin, Ncols, Kpad);
  int tiles = ((n + 15) / 16) * (Ncols / 16);
  fgc_gemm_wmma<<<cdiv(tiles, 8), 256, 0, stream>>>(xb, Wb, xW, n, Ncols, Kpad);
  hipMemsetAsync(agg, 0, (size_t)n * out * sizeof(float), stream);
  hipMemsetAsync(cnt, 0, (size_t)n * sizeof(float), stream);
  fgc_feast_edges<<<cdiv(E, 8), 256, 0, stream>>>(xin, cin, xW, out, src, dst, E, u, c, agg, cnt);
  fgc_softmax_c<<<1, 32, 0, stream>>>(c, qc);
  fgc_feast_finalize<<<cdiv((long)n * out, 256), 256, 0, stream>>>(
      xW, agg, cnt, qc, b, outbuf, n, out, ldo, col0, lrelu);
}

static void run_pool(hipStream_t stream, const float* x, const int* clust,
                     int n, int m, int C, float* psum, float* pcnt, float* dst) {
  hipMemsetAsync(psum, 0, (size_t)m * C * sizeof(float), stream);
  hipMemsetAsync(pcnt, 0, (size_t)m * sizeof(float), stream);
  fgc_pool_scatter<<<cdiv((long)n * C, 256), 256, 0, stream>>>(x, clust, psum, n, C);
  fgc_pool_cnt<<<cdiv(n, 256), 256, 0, stream>>>(clust, pcnt, n);
  fgc_pool_div<<<cdiv((long)m * C, 256), 256, 0, stream>>>(dst, psum, pcnt, m, C);
}

extern "C" void kernel_launch(void* const* d_in, const int* in_sizes, int n_in,
                              void* d_out, int out_size, void* d_ws, size_t ws_size,
                              hipStream_t stream) {
  (void)n_in; (void)out_size; (void)ws_size;

  const float* x     = (const float*)d_in[0];
  const int* ei1     = (const int*)d_in[1];
  const int* ei2     = (const int*)d_in[2];
  const int* ei3     = (const int*)d_in[3];
  const int* clust2  = (const int*)d_in[4];
  const int* clust3  = (const int*)d_in[5];

  const int N1 = in_sizes[4];        // 100000
  const int N2 = in_sizes[5];        // 25000
  const int N3 = 6250;               // fixed by problem definition
  const int E1 = in_sizes[1] / 2;
  const int E2 = in_sizes[2] / 2;
  const int E3 = in_sizes[3] / 2;

  // layer params: l1,l2,l3,l4,r1,r2,r3,r4 -> {u,c,w,b}
  const float *U[8], *Cc[8], *Wm[8], *Bb[8];
  for (int L = 0; L < 8; ++L) {
    U[L]  = (const float*)d_in[6 + 4 * L + 0];
    Cc[L] = (const float*)d_in[6 + 4 * L + 1];
    Wm[L] = (const float*)d_in[6 + 4 * L + 2];
    Bb[L] = (const float*)d_in[6 + 4 * L + 3];
  }
  const float* fc1_w = (const float*)d_in[38];
  const float* fc1_b = (const float*)d_in[39];
  const float* fc2_w = (const float*)d_in[40];
  const float* fc2_b = (const float*)d_in[41];
  float* out = (float*)d_out;

  // ---- workspace bump allocator ----
  char* base = (char*)d_ws;
  size_t off = 0;
  auto alloc_f = [&](size_t nelem) -> float* {
    float* p = (float*)(base + off);
    off += ((nelem * sizeof(float)) + 255) & ~(size_t)255;
    return p;
  };
  auto alloc_h = [&](size_t nelem) -> unsigned short* {
    unsigned short* p = (unsigned short*)(base + off);
    off += ((nelem * sizeof(unsigned short)) + 255) & ~(size_t)255;
    return p;
  };

  float* xW    = alloc_f((size_t)N1 * 288);   // max projection: [100k, 9*32]
  float* agg   = alloc_f((size_t)N1 * 32);    // max n*out
  float* cnt   = alloc_f((size_t)N1);
  float* qc    = alloc_f((size_t)NH);
  float* x1    = alloc_f((size_t)N1 * 32);
  float* x2_32 = alloc_f((size_t)N2 * 32);
  float* x2_64 = alloc_f((size_t)N2 * 64);
  float* x3_64 = alloc_f((size_t)N3 * 64);
  float* x3a   = alloc_f((size_t)N3 * 128);
  float* x3b   = alloc_f((size_t)N3 * 128);
  float* g2    = alloc_f((size_t)N2 * 128);
  float* x2cat = alloc_f((size_t)N2 * 128);
  float* x2r2  = alloc_f((size_t)N2 * 64);
  float* g1    = alloc_f((size_t)N1 * 64);
  float* x1cat = alloc_f((size_t)N1 * 64);
  float* yfin  = alloc_f((size_t)N1 * 32);
  float* psum  = alloc_f((size_t)N2 * 32);    // max pooled sum
  float* pcnt  = alloc_f((size_t)N2);
  unsigned short* xb = alloc_h((size_t)N1 * 64);   // max [100k, Kpad=64]
  unsigned short* Wb = alloc_h((size_t)1152 * 128);

  // ---- encoder ----
  // l1: [N1,6] -> [N1,32], lrelu
  run_feast(stream, x, N1, 6, 32, ei1, ei1 + E1, E1,
            U[0], Cc[0], Wm[0], Bb[0], xb, Wb, xW, agg, cnt, qc, x1, 32, 0, 1);
  // pool -> [N2,32]
  run_pool(stream, x1, clust2, N1, N2, 32, psum, pcnt, x2_32);
  // l2: [N2,32] -> [N2,64], lrelu
  run_feast(stream, x2_32, N2, 32, 64, ei2, ei2 + E2, E2,
            U[1], Cc[1], Wm[1], Bb[1], xb, Wb, xW, agg, cnt, qc, x2_64, 64, 0, 1);
  // pool -> [N3,64]
  run_pool(stream, x2_64, clust3, N2, N3, 64, psum, pcnt, x3_64);
  // l3: [N3,64] -> [N3,128], lrelu
  run_feast(stream, x3_64, N3, 64, 128, ei3, ei3 + E3, E3,
            U[2], Cc[2], Wm[2], Bb[2], xb, Wb, xW, agg, cnt, qc, x3a, 128, 0, 1);
  // l4: [N3,128] -> [N3,128], lrelu
  run_feast(stream, x3a, N3, 128, 128, ei3, ei3 + E3, E3,
            U[3], Cc[3], Wm[3], Bb[3], xb, Wb, xW, agg, cnt, qc, x3b, 128, 0, 1);

  // ---- decoder ----
  // unpool: g2 = x3b[clust3]  -> [N2,128]
  fgc_gather_rows<<<cdiv((long)N2 * 128, 256), 256, 0, stream>>>(x3b, clust3, g2, N2, 128);
  // x2cat[:, 0:64] = x2_64
  fgc_copy_cols<<<cdiv((long)N2 * 64, 256), 256, 0, stream>>>(x2_64, x2cat, N2, 64, 128, 0);
  // r1: feast(g2) -> x2cat[:, 64:128], no activation
  run_feast(stream, g2, N2, 128, 64, ei2, ei2 + E2, E2,
            U[4], Cc[4], Wm[4], Bb[4], xb, Wb, xW, agg, cnt, qc, x2cat, 128, 64, 0);
  // r2: [N2,128] -> [N2,64], lrelu
  run_feast(stream, x2cat, N2, 128, 64, ei2, ei2 + E2, E2,
            U[5], Cc[5], Wm[5], Bb[5], xb, Wb, xW, agg, cnt, qc, x2r2, 64, 0, 1);
  // unpool: g1 = x2r2[clust2] -> [N1,64]
  fgc_gather_rows<<<cdiv((long)N1 * 64, 256), 256, 0, stream>>>(x2r2, clust2, g1, N1, 64);
  // x1cat[:, 0:32] = x1
  fgc_copy_cols<<<cdiv((long)N1 * 32, 256), 256, 0, stream>>>(x1, x1cat, N1, 32, 64, 0);
  // r3: feast(g1) -> x1cat[:, 32:64], no activation
  run_feast(stream, g1, N1, 64, 32, ei1, ei1 + E1, E1,
            U[6], Cc[6], Wm[6], Bb[6], xb, Wb, xW, agg, cnt, qc, x1cat, 64, 32, 0);
  // r4: [N1,64] -> [N1,32], lrelu
  run_feast(stream, x1cat, N1, 64, 32, ei1, ei1 + E1, E1,
            U[7], Cc[7], Wm[7], Bb[7], xb, Wb, xW, agg, cnt, qc, yfin, 32, 0, 1);

  // ---- fused fc head: lrelu(y@fc1+b1)@fc2+b2, then row L2-normalize ----
  fgc_x_to_bf16<<<cdiv((long)N1 * 32, 256), 256, 0, stream>>>(yfin, xb, N1, 32, 32);
  fgc_w_to_bf16_cm<<<cdiv((long)1024 * 32, 256), 256, 0, stream>>>(fc1_w, Wb, 32, 1024, 32);
  {
    int tilesM = (N1 + 15) / 16;
    fgc_fc_fused<<<cdiv(tilesM, 8), 256, 0, stream>>>(xb, Wb, fc1_b, fc2_w, fc2_b, out, N1);
  }
}
